// IMCNet_50912542327435
// MI455X (gfx1250) — matrix operand order repaired and checked by reading, not compiled, not run on hardware
//
#include <hip/hip_runtime.h>
#include <math.h>

typedef __bf16 bf16;
typedef __attribute__((ext_vector_type(16))) __bf16 bf16x16;
typedef __attribute__((ext_vector_type(8)))  __bf16 bf16x8;
typedef __attribute__((ext_vector_type(8)))  float  v8f;

#define TB 256
#define BM 128
#define BN 64
#define KT 32
#define LDPAD 40   // LDS row stride (bf16 elems): 80B rows -> conflict-free b128 reads

__device__ __forceinline__ unsigned pack2bf(float a, float b) {
  unsigned ua = __builtin_bit_cast(unsigned, a);
  ua += 0x7fffu + ((ua >> 16) & 1u);               // RNE
  unsigned ub = __builtin_bit_cast(unsigned, b);
  ub += 0x7fffu + ((ub >> 16) & 1u);
  return (ua >> 16) | (ub & 0xffff0000u);
}

// ---------------------------------------------------------------------------
// Strided-batched GEMM: C[z] = alpha * A[z] * B[z] (+bias[row])
// A(row,k) = A + row*a_rs + k*a_cs ; B(k,col) = B + k*b_rs + col*b_cs
// C(row,col) = C + row*c_rs + col*c_cs
// per-z offsets: off = (z / zH) * *_z1 + (z % zH) * *_z2
// REQUIRES: M % 128 == 0, N % 64 == 0, K % 32 == 0 (no bounds guards).
// block = 8 waves (wave32), 128x64 C tile, 32 K-step, bf16 WMMA / f32 acc.
// ---------------------------------------------------------------------------
struct GemmParams {
  const float* A; const float* B; float* C; const float* bias;
  long a_rs, a_cs, a_z1, a_z2;
  long b_rs, b_cs, b_z1, b_z2;
  long c_rs, c_cs, c_z1, c_z2;
  int zH, M, N, K;
  float alpha;
};

__global__ __launch_bounds__(TB) void gemm_bf16_wmma(GemmParams p) {
  __shared__ alignas(16) bf16 sA[BM * LDPAD];
  __shared__ alignas(16) bf16 sB[BN * LDPAD];
  const int z  = blockIdx.z;
  const int z1 = z / p.zH, z2 = z % p.zH;
  const float* A = p.A + (long)z1 * p.a_z1 + (long)z2 * p.a_z2;
  const float* B = p.B + (long)z1 * p.b_z1 + (long)z2 * p.b_z2;
  float*       C = p.C + (long)z1 * p.c_z1 + (long)z2 * p.c_z2;
  const int m0b = blockIdx.y * BM;
  const int n0b = blockIdx.x * BN;
  const int t = threadIdx.x;
  const int lane = t & 31;
  const int wave = t >> 5;
  const int wr = wave >> 1, wc = wave & 1;        // 4x2 wave grid -> 128x64
  const int lr  = lane & 15;
  const int kbA = (lane < 16) ? 0 : 8;            // ISA 16-bit A layout
  const int kbB = (lane < 16) ? 0 : 8;            // ISA 16-bit B layout (same as A)

  // staging coordinates (affine in unroll index, hoisted out of K loop)
  const int rA  = t >> 4;                         // 0..15, + 16*i
  const int kpA = (t & 15) * 2;                   // even, fixed
  const int nB  = t & 63;                         // fixed
  const int kpB = (t >> 6) * 2;                   // 0..6, + 8*i
  const float* aPtr = A + (long)(m0b + rA) * p.a_rs + (long)kpA * p.a_cs;
  const float* bPtr = B + (long)(n0b + nB) * p.b_cs + (long)kpB * p.b_rs;
  const long aRow16 = 16L * p.a_rs;
  const long bK8    = 8L  * p.b_rs;

  v8f acc[2][2] = {};
  for (int k0 = 0; k0 < p.K; k0 += KT) {
    #pragma unroll
    for (int i = 0; i < 8; ++i) {                 // 128x32 A tile (2 bf16/thread/iter)
      const float* q = aPtr + i * aRow16;
      *reinterpret_cast<unsigned*>(&sA[(rA + i * 16) * LDPAD + kpA]) =
          pack2bf(q[0], q[p.a_cs]);
    }
    #pragma unroll
    for (int i = 0; i < 4; ++i) {                 // 32x64 B tile stored [n][k]
      const float* q = bPtr + i * bK8;
      *reinterpret_cast<unsigned*>(&sB[nB * LDPAD + kpB + i * 8]) =
          pack2bf(q[0], q[p.b_rs]);
    }
    __syncthreads();
    bf16x16 bfrag[2];
    #pragma unroll
    for (int j = 0; j < 2; ++j) {
      int n = wc * 32 + j * 16 + lr;
      bf16x8 lo = *reinterpret_cast<const bf16x8*>(&sB[n * LDPAD + kbB]);
      bf16x8 hi = *reinterpret_cast<const bf16x8*>(&sB[n * LDPAD + kbB + 16]);
      #pragma unroll
      for (int e = 0; e < 8; ++e) { bfrag[j][e] = lo[e]; bfrag[j][e + 8] = hi[e]; }
    }
    #pragma unroll
    for (int i = 0; i < 2; ++i) {
      int m = wr * 32 + i * 16 + lr;
      bf16x8 lo = *reinterpret_cast<const bf16x8*>(&sA[m * LDPAD + kbA]);
      bf16x8 hi = *reinterpret_cast<const bf16x8*>(&sA[m * LDPAD + kbA + 16]);
      bf16x16 afrag;
      #pragma unroll
      for (int e = 0; e < 8; ++e) { afrag[e] = lo[e]; afrag[e + 8] = hi[e]; }
      #pragma unroll
      for (int j = 0; j < 2; ++j) {
        acc[i][j] = __builtin_amdgcn_wmma_f32_16x16x32_bf16(
            false, afrag, false, bfrag[j], (short)0, acc[i][j], false, false);
      }
    }
    __syncthreads();
    aPtr += KT * p.a_cs;
    bPtr += KT * p.b_rs;
  }
  // ---------------- epilogue: one uniform bias branch, pointer-walk stores --
  const int mo = (lane < 16) ? 0 : 8;             // C/D layout: lane=N, vgpr r = M r/r+8
  float bloc[2][8];
  if (p.bias) {
    #pragma unroll
    for (int i = 0; i < 2; ++i)
      #pragma unroll
      for (int r = 0; r < 8; ++r)
        bloc[i][r] = p.bias[m0b + wr * 32 + i * 16 + mo + r];
  } else {
    #pragma unroll
    for (int i = 0; i < 2; ++i)
      #pragma unroll
      for (int r = 0; r < 8; ++r) bloc[i][r] = 0.f;
  }
  const long crs = p.c_rs;
  #pragma unroll
  for (int i = 0; i < 2; ++i)
    #pragma unroll
    for (int j = 0; j < 2; ++j) {
      const int gcol = n0b + wc * 32 + j * 16 + lr;
      float* cp = C + (long)(m0b + wr * 32 + i * 16 + mo) * crs + (long)gcol * p.c_cs;
      #pragma unroll
      for (int r = 0; r < 8; ++r) {
        *cp = p.alpha * acc[i][j][r] + bloc[i][r];
        cp += crs;
      }
    }
}

// --------------------------- softmax over rows -----------------------------
__global__ __launch_bounds__(TB) void softmax_rows(float* s, int rowlen) {
  float* p = s + (long)blockIdx.x * rowlen;
  __shared__ float red[TB];
  float mx = -INFINITY;
  for (int j = threadIdx.x; j < rowlen; j += TB) mx = fmaxf(mx, p[j]);
  red[threadIdx.x] = mx; __syncthreads();
  for (int o = TB / 2; o > 0; o >>= 1) {
    if (threadIdx.x < o) red[threadIdx.x] = fmaxf(red[threadIdx.x], red[threadIdx.x + o]);
    __syncthreads();
  }
  mx = red[0]; __syncthreads();
  float sum = 0.f;
  for (int j = threadIdx.x; j < rowlen; j += TB) { float e = __expf(p[j] - mx); p[j] = e; sum += e; }
  red[threadIdx.x] = sum; __syncthreads();
  for (int o = TB / 2; o > 0; o >>= 1) {
    if (threadIdx.x < o) red[threadIdx.x] += red[threadIdx.x + o];
    __syncthreads();
  }
  float inv = 1.0f / red[0];
  for (int j = threadIdx.x; j < rowlen; j += TB) p[j] *= inv;
}

// --------------------------- batch-norm ------------------------------------
__global__ __launch_bounds__(TB) void bn_stats(const float* tbuf, float* mean, float* rstd,
                                               int Npts, int Bb, long bstride) {
  int c = blockIdx.x;
  const float* base = tbuf + (long)c * Npts;
  float s = 0.f, s2 = 0.f;
  for (int b = 0; b < Bb; ++b)
    for (int j = threadIdx.x; j < Npts; j += TB) {
      float v = base[(long)b * bstride + j]; s += v; s2 += v * v;
    }
  __shared__ float r1[TB], r2[TB];
  r1[threadIdx.x] = s; r2[threadIdx.x] = s2; __syncthreads();
  for (int o = TB / 2; o > 0; o >>= 1) {
    if (threadIdx.x < o) { r1[threadIdx.x] += r1[threadIdx.x + o]; r2[threadIdx.x] += r2[threadIdx.x + o]; }
    __syncthreads();
  }
  if (threadIdx.x == 0) {
    float inv = 1.f / (float)(Bb * Npts);
    float m = r1[0] * inv;
    float var = r2[0] * inv - m * m;
    mean[c] = m; rstd[c] = rsqrtf(var + 1e-5f);
  }
}

__global__ void bn_apply_relu(float* tbuf, const float* mean, const float* rstd,
                              const float* gamma, const float* beta, long total, int Cc, int Npts) {
  long idx = (long)blockIdx.x * TB + threadIdx.x;
  if (idx >= total) return;
  int c = (int)((idx / Npts) % Cc);
  float v = (tbuf[idx] - mean[c]) * rstd[c] * gamma[c] + beta[c];
  tbuf[idx] = v > 0.f ? v : 0.f;
}

// --------------------------- elementwise helpers ---------------------------
__global__ void copy_f32(const float* s, float* d, long n) {
  long i = (long)blockIdx.x * TB + threadIdx.x; if (i < n) d[i] = s[i];
}
__global__ void add_inplace(float* d, const float* a, long n) {
  long i = (long)blockIdx.x * TB + threadIdx.x; if (i < n) d[i] += a[i];
}
__global__ void zero_f32(float* d, long n) {
  long i = (long)blockIdx.x * TB + threadIdx.x; if (i < n) d[i] = 0.f;
}
__global__ void copy_half_channels(const float* x, float* h, long DNl, long perb_out) {
  long i = (long)blockIdx.x * TB + threadIdx.x;
  if (i < 2 * DNl) { long b = i / DNl, r = i % DNl; h[b * perb_out + r] = x[i]; }
}

// --------------------------- Sinkhorn / OT ---------------------------------
__global__ void fill_bins(float* Z, const float* alpha, int n) {    // pitch n+1
  int i = blockIdx.x * TB + threadIdx.x;
  int b = blockIdx.y;
  if (i > n) return;
  float a = *alpha;
  float* Zb = Z + (long)b * (n + 1) * (n + 1);
  Zb[(long)i * (n + 1) + n] = a;   // last column
  Zb[(long)n * (n + 1) + i] = a;   // last row (i==n writes corner)
}

__global__ void transpose_sq(const float* src, float* dst, int n1) {
  __shared__ float tile[32][33];
  int bx = blockIdx.x * 32, by = blockIdx.y * 32, b = blockIdx.z;
  const float* S = src + (long)b * n1 * n1;
  float*       D = dst + (long)b * n1 * n1;
  int x = bx + threadIdx.x;
  for (int i = 0; i < 32; i += 8) {
    int y = by + threadIdx.y + i;
    if (x < n1 && y < n1) tile[threadIdx.y + i][threadIdx.x] = S[(long)y * n1 + x];
  }
  __syncthreads();
  int x2 = by + threadIdx.x;
  for (int i = 0; i < 32; i += 8) {
    int y2 = bx + threadIdx.y + i;
    if (x2 < n1 && y2 < n1) D[(long)y2 * n1 + x2] = tile[threadIdx.x][threadIdx.y + i];
  }
}

// w_out[b,i] = log_mu[i] - logsumexp_j( Zrows[b,i,j] + w_in[b,j] )
__global__ __launch_bounds__(TB) void sinkhorn_half(const float* Zrows, const float* w_in,
                                                    float* w_out, int n1) {
  const int i = blockIdx.x, b = blockIdx.y;
  const float* row = Zrows + ((long)b * n1 + i) * n1;
  const float* win = w_in + (long)b * n1;
  __shared__ float red[TB];
  float mx = -INFINITY;
  for (int j = threadIdx.x; j < n1; j += TB) mx = fmaxf(mx, row[j] + win[j]);
  red[threadIdx.x] = mx; __syncthreads();
  for (int o = TB / 2; o > 0; o >>= 1) {
    if (threadIdx.x < o) red[threadIdx.x] = fmaxf(red[threadIdx.x], red[threadIdx.x + o]);
    __syncthreads();
  }
  mx = red[0]; __syncthreads();
  float s = 0.f;
  for (int j = threadIdx.x; j < n1; j += TB) s += expf(row[j] + win[j] - mx);
  red[threadIdx.x] = s; __syncthreads();
  for (int o = TB / 2; o > 0; o >>= 1) {
    if (threadIdx.x < o) red[threadIdx.x] += red[threadIdx.x + o];
    __syncthreads();
  }
  if (threadIdx.x == 0) {
    const float kNorm = -7.6246189861593985f;   // -log(1024+1024)
    const float kLogN =  6.9314718055994531f;   // log(1024)
    float lse = mx + logf(red[0]);
    float lmu = (i == n1 - 1) ? (kLogN + kNorm) : kNorm;
    w_out[(long)b * n1 + i] = lmu - lse;
  }
}

__global__ void final_Z(float* Z, const float* u, const float* v, int n1) {
  long idx = (long)blockIdx.x * TB + threadIdx.x;
  long tot = (long)n1 * n1;
  if (idx >= tot) return;
  int b = blockIdx.y;
  long i = idx / n1, j = idx % n1;
  const float kNorm = -7.6246189861593985f;
  float* Zb = Z + (long)b * tot;
  Zb[idx] = Zb[idx] + u[(long)b * n1 + i] + v[(long)b * n1 + j] - kNorm;
}

// ---------------------------------------------------------------------------
extern "C" void kernel_launch(void* const* d_in, const int* in_sizes, int n_in,
                              void* d_out, int out_size, void* d_ws, size_t ws_size,
                              hipStream_t stream) {
  (void)in_sizes; (void)n_in; (void)out_size; (void)ws_size;
  const float* desc0 = (const float*)d_in[0];
  const float* desc1 = (const float*)d_in[1];
  const float* Wq = (const float*)d_in[2];  const float* bq = (const float*)d_in[3];
  const float* Wk = (const float*)d_in[4];  const float* bk = (const float*)d_in[5];
  const float* Wv = (const float*)d_in[6];  const float* bv = (const float*)d_in[7];
  const float* Wm = (const float*)d_in[8];  const float* bm = (const float*)d_in[9];
  const float* W1 = (const float*)d_in[10]; const float* b1 = (const float*)d_in[11];
  const float* gam = (const float*)d_in[12]; const float* bet = (const float*)d_in[13];
  const float* W2 = (const float*)d_in[14]; const float* b2 = (const float*)d_in[15];
  const float* Wf = (const float*)d_in[16]; const float* bfv = (const float*)d_in[17];
  const float* bin_score = (const float*)d_in[18];

  const int  Bb = 2, Dd = 256, Nn = 1024, Hh = 4, C2 = 512, Lnum = 18;
  const long DN = (long)Dd * Nn, BDN = (long)Bb * DN;
  const long NN = (long)Nn * Nn, HNN = (long)Hh * NN;
  const int  n1 = Nn + 1;

  float* ws = (float*)d_ws;
  size_t off = 0;
  auto alloc = [&](size_t n) { float* p = ws + off; off += n; return p; };
  float* dx0  = alloc(BDN);  float* dx1  = alloc(BDN);
  float* del0 = alloc(BDN);  float* del1 = alloc(BDN);
  float* qb   = alloc(BDN);  float* kbuf = alloc(BDN);  float* vbuf2 = alloc(BDN);
  float* sc   = alloc((size_t)Bb * HNN);
  float* ao   = alloc(BDN);
  float* hb   = alloc((size_t)Bb * C2 * Nn);
  float* tb   = alloc((size_t)Bb * C2 * Nn);
  float* mean = alloc(C2);   float* rstd = alloc(C2);
  float* Z0T  = alloc((size_t)Bb * n1 * n1);
  float* ub   = alloc((size_t)Bb * n1);
  float* vb_  = alloc((size_t)Bb * n1);

  float* gout0 = (float*)d_out;
  float* gout1 = gout0 + BDN;
  float* Z0    = gout1 + BDN;

  auto launch_gemm = [&](GemmParams p, int Z) {
    dim3 g((unsigned)(p.N / BN), (unsigned)(p.M / BM), (unsigned)Z);
    gemm_bf16_wmma<<<g, dim3(TB), 0, stream>>>(p);
  };
  // conv1x1: C[b,M,N] = W[M,K] * X[b,K,N] + bias
  auto conv = [&](const float* Wt, const float* bias, const float* X, float* Cout,
                  int M, int K, long c_z1) {
    GemmParams p{};
    p.A = Wt;   p.a_rs = K;  p.a_cs = 1; p.a_z1 = 0;            p.a_z2 = 0;
    p.B = X;    p.b_rs = Nn; p.b_cs = 1; p.b_z1 = (long)K * Nn; p.b_z2 = 0;
    p.C = Cout; p.c_rs = Nn; p.c_cs = 1; p.c_z1 = c_z1;         p.c_z2 = 0;
    p.bias = bias; p.zH = 1; p.M = M; p.N = Nn; p.K = K; p.alpha = 1.f;
    launch_gemm(p, Bb);
  };

  const int gBDN = (int)((BDN + TB - 1) / TB);
  copy_f32<<<gBDN, TB, 0, stream>>>(desc0, dx0, BDN);
  copy_f32<<<gBDN, TB, 0, stream>>>(desc1, dx1, BDN);

  for (int l = 0; l < Lnum; ++l) {
    const bool cross = (l & 1);
    const float* Wq_l = Wq + (long)l * Dd * Dd;  const float* bq_l = bq + (long)l * Dd;
    const float* Wk_l = Wk + (long)l * Dd * Dd;  const float* bk_l = bk + (long)l * Dd;
    const float* Wv_l = Wv + (long)l * Dd * Dd;  const float* bv_l = bv + (long)l * Dd;
    const float* Wm_l = Wm + (long)l * Dd * Dd;  const float* bm_l = bm + (long)l * Dd;
    const float* W1_l = W1 + (long)l * C2 * C2;  const float* b1_l = b1 + (long)l * C2;
    const float* W2_l = W2 + (long)l * Dd * C2;  const float* b2_l = b2 + (long)l * Dd;
    const float* g_l  = gam + (long)l * C2;      const float* be_l = bet + (long)l * C2;

    for (int path = 0; path < 2; ++path) {
      const float* x   = path ? dx1 : dx0;
      const float* src = cross ? (path ? dx0 : dx1) : x;
      float* del = path ? del1 : del0;

      conv(Wq_l, bq_l, x,   qb,    Dd, Dd, DN);
      conv(Wk_l, bk_l, src, kbuf,  Dd, Dd, DN);
      conv(Wv_l, bv_l, src, vbuf2, Dd, Dd, DN);

      { // scores[b,h,n,m] = sum_dh q[b,dh*4+h,n] * k[b,dh*4+h,m] / 8
        GemmParams p{};
        p.A = qb;    p.a_rs = 1;        p.a_cs = 4L * Nn; p.a_z1 = DN;  p.a_z2 = Nn;
        p.B = kbuf;  p.b_rs = 4L * Nn;  p.b_cs = 1;       p.b_z1 = DN;  p.b_z2 = Nn;
        p.C = sc;    p.c_rs = Nn;       p.c_cs = 1;       p.c_z1 = HNN; p.c_z2 = NN;
        p.bias = nullptr; p.zH = Hh; p.M = Nn; p.N = Nn; p.K = 64;
        p.alpha = 0.125f;
        launch_gemm(p, Bb * Hh);
      }
      softmax_rows<<<dim3((unsigned)(Bb * Hh * Nn)), TB, 0, stream>>>(sc, Nn);
      { // transposed attn*V: out^T(n,dh) = sum_m P(n,m) V(dh,m)  [M=1024,N=64,K=1024]
        GemmParams p{};
        p.A = sc;    p.a_rs = Nn;      p.a_cs = 1;       p.a_z1 = HNN; p.a_z2 = NN;
        p.B = vbuf2; p.b_rs = 1;       p.b_cs = 4L * Nn; p.b_z1 = DN;  p.b_z2 = Nn;
        p.C = ao;    p.c_rs = 1;       p.c_cs = 4L * Nn; p.c_z1 = DN;  p.c_z2 = Nn;
        p.bias = nullptr; p.zH = Hh; p.M = Nn; p.N = 64; p.K = Nn;
        p.alpha = 1.f;
        launch_gemm(p, Bb * Hh);
      }
      // msg -> h[:,256:512,:] ; x -> h[:,0:256,:]
      conv(Wm_l, bm_l, ao, hb + (long)Dd * Nn, Dd, Dd, (long)C2 * Nn);
      copy_half_channels<<<gBDN, TB, 0, stream>>>(x, hb, DN, (long)C2 * Nn);
      // t = W1*h + b1 ; BN(train stats) ; relu ; delta = W2*t + b2
      conv(W1_l, b1_l, hb, tb, C2, C2, (long)C2 * Nn);
      bn_stats<<<C2, TB, 0, stream>>>(tb, mean, rstd, Nn, Bb, (long)C2 * Nn);
      bn_apply_relu<<<(int)(((long)Bb * C2 * Nn + TB - 1) / TB), TB, 0, stream>>>(
          tb, mean, rstd, g_l, be_l, (long)Bb * C2 * Nn, C2, Nn);
      conv(W2_l, b2_l, tb, del, Dd, C2, DN);
    }
    add_inplace<<<gBDN, TB, 0, stream>>>(dx0, del0, BDN);
    add_inplace<<<gBDN, TB, 0, stream>>>(dx1, del1, BDN);
  }

  // final projections straight into d_out
  conv(Wf, bfv, dx0, gout0, Dd, Dd, DN);
  conv(Wf, bfv, dx1, gout1, Dd, Dd, DN);

  { // scores[b,n,m] = sum_d g0[b,d,n]*g1[b,d,m] / 16, pitched into 1025x1025 Z0
    GemmParams p{};
    p.A = gout0; p.a_rs = 1;  p.a_cs = Nn; p.a_z1 = DN; p.a_z2 = 0;
    p.B = gout1; p.b_rs = Nn; p.b_cs = 1;  p.b_z1 = DN; p.b_z2 = 0;
    p.C = Z0;    p.c_rs = n1; p.c_cs = 1;  p.c_z1 = (long)n1 * n1; p.c_z2 = 0;
    p.bias = nullptr; p.zH = 1; p.M = Nn; p.N = Nn; p.K = Dd;
    p.alpha = 0.0625f;
    launch_gemm(p, Bb);
  }
  fill_bins<<<dim3((n1 + TB - 1) / TB, Bb), TB, 0, stream>>>(Z0, bin_score, Nn);
  transpose_sq<<<dim3((n1 + 31) / 32, (n1 + 31) / 32, Bb), dim3(32, 8), 0, stream>>>(Z0, Z0T, n1);
  zero_f32<<<(int)((2L * Bb * n1 + TB - 1) / TB), TB, 0, stream>>>(ub, 2L * Bb * n1); // u & v contiguous

  for (int it = 0; it < 100; ++it) {
    sinkhorn_half<<<dim3(n1, Bb), TB, 0, stream>>>(Z0,  vb_, ub,  n1);  // u update (rows)
    sinkhorn_half<<<dim3(n1, Bb), TB, 0, stream>>>(Z0T, ub,  vb_, n1);  // v update (cols via Z0T)
  }
  final_Z<<<dim3((unsigned)(((long)n1 * n1 + TB - 1) / TB), Bb), TB, 0, stream>>>(Z0, ub, vb_, n1);
}
